// RNN_torch_63264868270750
// MI455X (gfx1250) — matrix operand order repaired and checked by reading, not compiled, not run on hardware
//
#include <hip/hip_runtime.h>
#include <hip/hip_bf16.h>

#define RN_N   512
#define RN_B   128
#define RN_T   1000
#define RN_INP 6
#define RN_ALPHA 0.1f   // DT/TAU = 1.0/10.0

typedef __attribute__((ext_vector_type(16))) __bf16 v16bf;
typedef __attribute__((ext_vector_type(8)))  float  v8f;

// ---------------------------------------------------------------------------
// Pack W_rec (f32, row-major [n][k]) into bf16 WMMA B-fragments.
// B-matrix (32x16 bf16) VGPR layout (ISA 7.12.2 / sparse B layout):
//   lanes 0-15 hold K=0..7,16..23 for column N=lane, lanes 16-31 hold K=8..15,24..31.
//   Per lane, 16 bf16 "slots" (8 VGPRs x 2 halves) hold K=8h+s (s<8), K=8+8h+s (s>=8), h=lane/16.
// Fragment f = ntile*16 + kblk, 512 bf16 each; lane data contiguous (32 B).
// ---------------------------------------------------------------------------
__global__ void pack_wrec_kernel(const float* __restrict__ wrec,
                                 __bf16* __restrict__ pk) {
    int idx = blockIdx.x * 256 + threadIdx.x;       // 0 .. N*N-1
    int n = idx >> 9;                               // row of W_rec
    int k = idx & (RN_N - 1);                       // col of W_rec
    float v = wrec[idx];                            // B[k][n] = W_rec[n][k]
    int ntile = n >> 4, nl = n & 15;
    int kblk  = k >> 5, kk = k & 31;
    int lane  = (((kk >> 3) & 1) << 4) | nl;
    int slot  = ((kk >> 4) << 3) + (kk & 7);
    pk[(((size_t)ntile * 16 + kblk) << 9) + (lane << 4) + slot] = (__bf16)v;
}

// A-pack LDS index: A-matrix (16x32 bf16) VGPR layout (ISA 7.12.2):
//   k-group g = (k%32)/8 ; lanes 0-15 hold g=0,2 ; lanes 16-31 hold g=1,3
//   slots 0-7 hold g<2 K's, slots 8-15 hold g>=2 K's.
__device__ __forceinline__ int apack_idx(int buf, int m, int k) {
    int kblk = k >> 5, kk = k & 31, g = kk >> 3;
    int la   = m + ((g & 1) << 4);
    int slot = ((g >> 1) << 3) + (kk & 7);
    return (buf << 13) + (kblk << 9) + (la << 4) + slot;
}

// ---------------------------------------------------------------------------
// Persistent RNN kernel: 8 blocks x 512 threads (16 waves of 32).
// Block bg owns batch rows [bg*16, bg*16+16); wave w owns N-tiles 2w, 2w+1.
// State kept in f32 registers (v8f per tile, D-tile layout), re-quantized to
// bf16 into a double-buffered LDS A-pack each step.
// ---------------------------------------------------------------------------
__launch_bounds__(512, 1)
__global__ void rnn_dynamics_kernel(const float* __restrict__ u,
                                    const float* __restrict__ rnoise,
                                    const float* __restrict__ inoise,
                                    const __bf16* __restrict__ wpk,
                                    const float* __restrict__ winp,
                                    const float* __restrict__ yinit,
                                    float* __restrict__ states) {
    __shared__ __align__(32) __bf16 lds_a[2 * 16 * 512];    // 32 KB, 2 bufs
    __shared__ float lds_winp[RN_N * RN_INP];               // 12 KB
    __shared__ float lds_uin[16 * RN_INP];

    const int tid  = threadIdx.x;
    const int lane = tid & 31;
    const int wave = tid >> 5;          // 0..15
    const int bg   = blockIdx.x;        // 0..7
    const int hi   = lane >> 4;         // half-wave: rows 0-7 vs 8-15
    const int nn   = lane & 15;         // column within tile

    // Preload W_inp [n][j] into LDS (512 threads -> one row each).
    #pragma unroll
    for (int j = 0; j < RN_INP; ++j)
        lds_winp[tid * RN_INP + j] = winp[tid * RN_INP + j];

    const int nt0 = wave * 2, nt1 = wave * 2 + 1;
    const int n0 = nt0 * 16 + nn, n1 = nt1 * 16 + nn;

    // Init state from y_init (broadcast over batch rows).
    v8f s0v, s1v;
    const float y0 = yinit[n0], y1 = yinit[n1];
    #pragma unroll
    for (int r = 0; r < 8; ++r) { s0v[r] = y0; s1v[r] = y1; }

    // states[:,0,:] = y_init ; initial A-pack in buffer 0.
    #pragma unroll
    for (int r = 0; r < 8; ++r) {
        int m = r + hi * 8;
        size_t b = (size_t)bg * 16 + m;
        states[(b * RN_T) * RN_N + n0] = y0;
        states[(b * RN_T) * RN_N + n1] = y1;
        lds_a[apack_idx(0, m, n0)] = (__bf16)y0;
        lds_a[apack_idx(0, m, n1)] = (__bf16)y1;
    }

    // Per-lane B fragment base addresses (L2-resident packed W_rec).
    const __bf16* bbase0 = wpk + ((size_t)(nt0 * 16) << 9) + (lane << 4);
    const __bf16* bbase1 = wpk + ((size_t)(nt1 * 16) << 9) + (lane << 4);

    for (int t = 0; t < RN_T - 1; ++t) {
        const int cur = t & 1, nxt = cur ^ 1;

        // Stage (u + inp_noise)[b, t, :] for this block's 16 rows.
        if (tid < 16 * RN_INP) {
            int m = tid / RN_INP, j = tid - m * RN_INP;
            size_t off = ((size_t)(bg * 16 + m) * RN_T + t) * RN_INP + j;
            lds_uin[tid] = u[off] + inoise[off];
        }
        __syncthreads();   // uin ready; A-pack[cur] ready from previous step

        // C-init = input term: (u+in)[m,:] . W_inp[n,:]
        float w0[RN_INP], w1[RN_INP];
        #pragma unroll
        for (int j = 0; j < RN_INP; ++j) {
            w0[j] = lds_winp[n0 * RN_INP + j];
            w1[j] = lds_winp[n1 * RN_INP + j];
        }
        v8f acc0, acc1;
        #pragma unroll
        for (int r = 0; r < 8; ++r) {
            int m = r + hi * 8;
            float a0 = 0.f, a1 = 0.f;
            #pragma unroll
            for (int j = 0; j < RN_INP; ++j) {
                float uv = lds_uin[m * RN_INP + j];
                a0 += uv * w0[j];
                a1 += uv * w1[j];
            }
            acc0[r] = a0; acc1[r] = a1;
        }

        // Launder a zero *offset* (not the pointer!) each iteration: keeps the
        // address chain rooted at the global kernarg pointer (so loads lower
        // to global_load_b128, not flat_load_b128), while the unknown offset
        // defeats LICM so fragment loads are re-issued every step (no spills).
        int lz = 0;
        asm volatile("" : "+v"(lz));
        const v16bf* Bf0 = (const v16bf*)(bbase0 + lz);
        const v16bf* Bf1 = (const v16bf*)(bbase1 + lz);

        const v16bf* Af = (const v16bf*)&lds_a[(cur << 13) + (lane << 4)];

        // K = 512 reduction, software-pipelined:
        // load fragments for kb+1 while the two WMMAs for kb issue.
        v16bf a  = Af[0];
        v16bf b0 = Bf0[0];
        v16bf b1 = Bf1[0];
        #pragma unroll
        for (int kb = 0; kb < 16; ++kb) {
            v16bf an, b0n, b1n;
            if (kb < 15) {
                an  = Af[(kb + 1) * 32];    // +1024 B per kblk
                b0n = Bf0[(kb + 1) * 32];
                b1n = Bf1[(kb + 1) * 32];
            }
            acc0 = __builtin_amdgcn_wmma_f32_16x16x32_bf16(
                       false, a, false, b0, (short)0, acc0, false, false);
            acc1 = __builtin_amdgcn_wmma_f32_16x16x32_bf16(
                       false, a, false, b1, (short)0, acc1, false, false);
            if (kb < 15) { a = an; b0 = b0n; b1 = b1n; }
        }

        // Leaky-ReLU update; emit f32 states and bf16 next A-pack.
        #pragma unroll
        for (int r = 0; r < 8; ++r) {
            int m = r + hi * 8;
            size_t row = (size_t)(bg * 16 + m) * RN_T + t;
            float rn0 = rnoise[row * RN_N + n0];
            float rn1 = rnoise[row * RN_N + n1];
            float p0 = acc0[r]; p0 = p0 > 0.f ? p0 : 0.f;
            float p1 = acc1[r]; p1 = p1 > 0.f ? p1 : 0.f;
            float sn0 = (1.f - RN_ALPHA) * s0v[r] + RN_ALPHA * (p0 + rn0);
            float sn1 = (1.f - RN_ALPHA) * s1v[r] + RN_ALPHA * (p1 + rn1);
            s0v[r] = sn0; s1v[r] = sn1;
            states[(row + 1) * RN_N + n0] = sn0;   // (b, t+1, n)
            states[(row + 1) * RN_N + n1] = sn1;
            lds_a[apack_idx(nxt, m, n0)] = (__bf16)sn0;
            lds_a[apack_idx(nxt, m, n1)] = (__bf16)sn1;
        }
        __syncthreads();   // next A-pack complete; uin consumers done
    }
}

// ---------------------------------------------------------------------------
// Readout: out[b,t] = states[b,t,:] . W_out   (one wave per row)
// ---------------------------------------------------------------------------
__global__ void readout_kernel(const float* __restrict__ states,
                               const float* __restrict__ wout,
                               float* __restrict__ out) {
    __shared__ float lw[RN_N];
    int tid = threadIdx.x;
    for (int i = tid; i < RN_N; i += 256) lw[i] = wout[i];
    __syncthreads();
    int lane = tid & 31, wave = tid >> 5;
    size_t row = (size_t)blockIdx.x * 8 + wave;
    if (row < (size_t)RN_B * RN_T) {
        const float* sp = states + row * RN_N;
        float sum = 0.f;
        #pragma unroll
        for (int i = 0; i < 16; ++i) {
            int n = lane + i * 32;
            sum += sp[n] * lw[n];
        }
        #pragma unroll
        for (int off = 16; off > 0; off >>= 1)
            sum += __shfl_down(sum, off, 32);
        if (lane == 0) out[row] = sum;
    }
}

extern "C" void kernel_launch(void* const* d_in, const int* in_sizes, int n_in,
                              void* d_out, int out_size, void* d_ws, size_t ws_size,
                              hipStream_t stream) {
    const float* u     = (const float*)d_in[0];
    const float* rnois = (const float*)d_in[1];
    const float* inois = (const float*)d_in[2];
    const float* wrec  = (const float*)d_in[3];
    const float* winp  = (const float*)d_in[4];
    const float* wout  = (const float*)d_in[5];
    const float* yini  = (const float*)d_in[6];

    float* states = (float*)d_out;                         // (B, T, N)
    float* outs   = states + (size_t)RN_B * RN_T * RN_N;   // (B, T, 1)
    __bf16* wpk   = (__bf16*)d_ws;                         // 512 KB packed W_rec

    hipLaunchKernelGGL(pack_wrec_kernel,
                       dim3((RN_N * RN_N) / 256), dim3(256), 0, stream,
                       wrec, wpk);
    hipLaunchKernelGGL(rnn_dynamics_kernel,
                       dim3(RN_B / 16), dim3(512), 0, stream,
                       u, rnois, inois, wpk, winp, yini, states);
    hipLaunchKernelGGL(readout_kernel,
                       dim3((RN_B * RN_T + 7) / 8), dim3(256), 0, stream,
                       states, wout, outs);
}
